// MyMap0Layer_31207232372879
// MI455X (gfx1250) — compile-verified, matrix operand order checked by reading, not run on hardware
//
#include <hip/hip_runtime.h>
#include <math.h>

// Problem constants (match reference)
#define HDIM 2048
#define IDIM 5632
#define BATCH 2
#define SEQ 2048
#define NTOK (BATCH * SEQ)
#define EPSV 1e-5f

typedef __attribute__((ext_vector_type(16))) _Float16 v16h;
typedef __attribute__((ext_vector_type(8)))  _Float16 v8h;
typedef __attribute__((ext_vector_type(8)))  float    v8f;
typedef int vint4 __attribute__((vector_size(16)));   // matches async-LDS builtin pointee

union FragU { v16h v; v8h h[2]; };

// GEMM tiling: 128x128 block tile, K-step 32, 8 waves (wave32) in a 4(M) x 2(N) grid,
// each wave owns a 32x64 subtile = 2x4 WMMA 16x16 fragments. Double-buffered LDS.
#define BM 128
#define BN 128
#define BKT 32
#define LDST 40   // padded LDS row stride in halves (80 B, keeps 16-B alignment, spreads banks)

constexpr int EP_F16 = 0, EP_F32 = 1, EP_SILU = 2, EP_CAUSAL = 3, EP_TRANS = 4;

// CDNA5 async global->LDS copy path (ASYNCcnt-tracked). Guarded so the file
// still compiles on toolchains without the builtins (falls back to VGPR staging).
#if defined(__has_builtin)
#if __has_builtin(__builtin_amdgcn_global_load_async_to_lds_b128) && \
    __has_builtin(__builtin_amdgcn_s_wait_asynccnt)
#define USE_ASYNC_LDS 1
#endif
#endif

#if defined(USE_ASYNC_LDS)
__device__ __forceinline__ void cp16_async(const _Float16* g, _Float16* l) {
  __builtin_amdgcn_global_load_async_to_lds_b128(
      (__attribute__((address_space(1))) vint4*)
          (__attribute__((address_space(1))) void*)g,
      (__attribute__((address_space(3))) vint4*)
          (__attribute__((address_space(3))) void*)l,
      0, 0);
}
#endif

// Load one 16x32 f16 WMMA operand fragment from an LDS tile (row-major, stride LDST).
// ISA 7.12.2 16-bit A layout: lane L (L&15 = row), k-base = (L&16 ? 8 : 0),
// VGPRs 0..3 hold K kb..kb+7, VGPRs 4..7 hold K kb+16..kb+23  -> two b128 LDS loads.
__device__ __forceinline__ v16h ld_frag(const _Float16* p, int lane) {
  const _Float16* q = p + (lane & 15) * LDST + ((lane & 16) ? 8 : 0);
  FragU f;
  f.h[0] = *(const v8h*)(q);
  f.h[1] = *(const v8h*)(q + 16);
  return f.v;
}

// C = act( A[M,K](f16) @ W[N,K](f16)^T * scale )
// grid: (N/BN, M/BM, Z) with per-z element strides sAz/sWz/sCz/sEz.
template<int EP>
__global__ __launch_bounds__(256)
void gemm_wmma(const _Float16* __restrict__ A, const _Float16* __restrict__ W,
               void* __restrict__ Cp, const float* __restrict__ E,
               int M, int N, int K, float scale,
               long long sAz, long long sWz, long long sCz, long long sEz,
               int causalSkip)
{
  // [buffer][A=0/B=1][tile]
  __shared__ __align__(16) _Float16 ls[2][2][BM * LDST];

  const int tid  = threadIdx.x;
  const int lane = tid & 31;
  const int wave = tid >> 5;
  const int wm   = wave & 3;     // wave row in M (0..3) -> 32 rows each
  const int wn   = wave >> 2;    // wave col in N (0..1) -> 64 cols each
  const int bm   = blockIdx.y * BM;
  const int bn   = blockIdx.x * BN;
  const int z    = blockIdx.z;

  const _Float16* Ab = A + (long long)z * sAz;
  const _Float16* Wb = W + (long long)z * sWz;

  // Per-thread staging chunks: 512 16-B chunks per 128x32 tile, 2 per thread per operand.
  const int c0   = tid * 2;
  const int rA0  = c0 >> 2,       cA0 = (c0 & 3) * 8;
  const int rA1  = (c0 + 1) >> 2, cA1 = ((c0 + 1) & 3) * 8;

  v8f acc[2][4] = {};   // 8 f32 C fragments per wave, zero-initialized

  const bool skip = (causalSkip != 0) && (bm + BM <= bn);  // fully above causal diagonal

  auto stage = [&](int buf, int k0) {
#if defined(USE_ASYNC_LDS)
    // 4 async-to-LDS instructions per wave per tile -> ASYNCcnt discipline below.
    cp16_async(Ab + (long long)(bm + rA0) * K + k0 + cA0, &ls[buf][0][rA0 * LDST + cA0]);
    cp16_async(Ab + (long long)(bm + rA1) * K + k0 + cA1, &ls[buf][0][rA1 * LDST + cA1]);
    cp16_async(Wb + (long long)(bn + rA0) * K + k0 + cA0, &ls[buf][1][rA0 * LDST + cA0]);
    cp16_async(Wb + (long long)(bn + rA1) * K + k0 + cA1, &ls[buf][1][rA1 * LDST + cA1]);
#else
    v8h a0 = *(const v8h*)(Ab + (long long)(bm + rA0) * K + k0 + cA0);
    v8h a1 = *(const v8h*)(Ab + (long long)(bm + rA1) * K + k0 + cA1);
    v8h b0 = *(const v8h*)(Wb + (long long)(bn + rA0) * K + k0 + cA0);
    v8h b1 = *(const v8h*)(Wb + (long long)(bn + rA1) * K + k0 + cA1);
    *(v8h*)(&ls[buf][0][rA0 * LDST + cA0]) = a0;
    *(v8h*)(&ls[buf][0][rA1 * LDST + cA1]) = a1;
    *(v8h*)(&ls[buf][1][rA0 * LDST + cA0]) = b0;
    *(v8h*)(&ls[buf][1][rA1 * LDST + cA1]) = b1;
#endif
  };

  if (!skip) {
    const int T = K / BKT;
    stage(0, 0);                       // prologue fill of buffer 0
    for (int t = 0; t < T; ++t) {
      const int cur = t & 1;
      const bool more = (t + 1) < T;
      if (more) stage(cur ^ 1, (t + 1) * BKT);   // overlap next tile with compute
#if defined(USE_ASYNC_LDS)
      // Async loads complete in order per wave: <=4 outstanding means tile t landed.
      if (more) __builtin_amdgcn_s_wait_asynccnt(4);
      else      __builtin_amdgcn_s_wait_asynccnt(0);
#endif
      __syncthreads();                 // publish tile t to all waves

      v16h af[2], bf[4];
      #pragma unroll
      for (int mi = 0; mi < 2; ++mi)
        af[mi] = ld_frag(&ls[cur][0][(wm * 32 + mi * 16) * LDST], lane);
      #pragma unroll
      for (int ni = 0; ni < 4; ++ni)
        bf[ni] = ld_frag(&ls[cur][1][(wn * 64 + ni * 16) * LDST], lane);
      #pragma unroll
      for (int mi = 0; mi < 2; ++mi)
        #pragma unroll
        for (int ni = 0; ni < 4; ++ni)
          acc[mi][ni] = __builtin_amdgcn_wmma_f32_16x16x32_f16(
              false, af[mi], false, bf[ni], (short)0, acc[mi][ni], false, false);
      __syncthreads();                 // buffer `cur` free for restage at t+2
    }
  }

  // Epilogue. C/D layout: VGPR r -> row r (lanes 16..31: +8), col = lane&15.
  _Float16* Ch = (_Float16*)Cp + (long long)z * sCz;
  float*    Cf = (float*)Cp    + (long long)z * sCz;
  const float* Ef = E ? (E + (long long)z * sEz) : nullptr;
  const int nn0 = bn + wn * 64;
  const int mm0 = bm + wm * 32 + ((lane & 16) ? 8 : 0);
  const int nl  = lane & 15;
  #pragma unroll
  for (int mi = 0; mi < 2; ++mi) {
    #pragma unroll
    for (int ni = 0; ni < 4; ++ni) {
      int n = nn0 + ni * 16 + nl;
      #pragma unroll
      for (int r = 0; r < 8; ++r) {
        int m = mm0 + mi * 16 + r;
        float v = acc[mi][ni][r] * scale;
        long long idx = (long long)m * N + n;
        if constexpr (EP == EP_F16) {
          Ch[idx] = (_Float16)v;
        } else if constexpr (EP == EP_F32) {
          Cf[idx] = v;
        } else if constexpr (EP == EP_SILU) {           // Hmid = silu(Graw) * U
          float e = Ef[idx];
          Ch[idx] = (_Float16)((e / (1.f + __expf(-e))) * v);
        } else if constexpr (EP == EP_CAUSAL) {         // P = causal * silu(S1) * S2
          float o = 0.f;
          if (m >= n) { float e = Ef[idx]; o = (e / (1.f + __expf(-e))) * v; }
          Ch[idx] = (_Float16)o;
        } else {                                        // EP_TRANS: Vt[b][n][s]
          int b = m >> 11, s = m & (SEQ - 1);
          Ch[((long long)b * N + n) * SEQ + s] = (_Float16)v;
        }
      }
    }
  }
}

// fp32 -> f16 weight conversion (one-shot, ~205 MB total => ~9 us at 23.3 TB/s)
__global__ __launch_bounds__(256)
void cvt_f32_f16(const float* __restrict__ s, _Float16* __restrict__ d, long long n4)
{
  long long i = (long long)blockIdx.x * 256 + threadIdx.x;
  if (i < n4) {
    float4 v = ((const float4*)s)[i];
    union { _Float16 h[4]; unsigned long long u; } t;
    t.h[0] = (_Float16)v.x; t.h[1] = (_Float16)v.y;
    t.h[2] = (_Float16)v.z; t.h[3] = (_Float16)v.w;
    ((unsigned long long*)d)[i] = t.u;
  }
}

// RMSNorm + write Xn (f16) + RoPE'd query Q (f16). One block per token.
__global__ __launch_bounds__(256)
void rmsnorm_rope_q(const float* __restrict__ X, const long long* __restrict__ pos,
                    const float* __restrict__ wln,
                    _Float16* __restrict__ Xn, _Float16* __restrict__ Q)
{
  __shared__ float sh[HDIM];
  __shared__ float red[8];
  __shared__ float srstd;
  const int t = blockIdx.x;
  const float* x = X + (long long)t * HDIM;
  float ss = 0.f;
  for (int i = threadIdx.x; i < HDIM; i += 256) { float v = x[i]; sh[i] = v; ss += v * v; }
  for (int o = 16; o > 0; o >>= 1) ss += __shfl_down(ss, o, 32);   // wave32
  if ((threadIdx.x & 31) == 0) red[threadIdx.x >> 5] = ss;
  __syncthreads();
  if (threadIdx.x == 0) {
    float s = 0.f;
    #pragma unroll
    for (int i = 0; i < 8; ++i) s += red[i];
    srstd = rsqrtf(s / (float)HDIM + EPSV);
  }
  __syncthreads();
  float rstd = srstd;
  for (int i = threadIdx.x; i < HDIM; i += 256) {
    float nv = sh[i] * rstd * wln[i];
    sh[i] = nv;
    Xn[(long long)t * HDIM + i] = (_Float16)nv;
  }
  __syncthreads();
  float p = (float)pos[t];
  for (int j = threadIdx.x; j < HDIM / 2; j += 256) {
    float inv = __powf(10000.f, -(float)(2 * j) / (float)HDIM);
    float ang = p * inv;
    float s, c; __sincosf(ang, &s, &c);
    float x0 = sh[j], x1 = sh[j + HDIM / 2];
    Q[(long long)t * HDIM + j]            = (_Float16)(x0 * c - x1 * s);
    Q[(long long)t * HDIM + j + HDIM / 2] = (_Float16)(x1 * c + x0 * s);
  }
}

// In-place RoPE on K1 (f16). One block per token; each thread owns disjoint (j, j+H/2) pairs.
__global__ __launch_bounds__(256)
void rope_k1(_Float16* __restrict__ K1, const long long* __restrict__ pos)
{
  const int t = blockIdx.x;
  float p = (float)pos[t];
  _Float16* row = K1 + (long long)t * HDIM;
  for (int j = threadIdx.x; j < HDIM / 2; j += 256) {
    float inv = __powf(10000.f, -(float)(2 * j) / (float)HDIM);
    float ang = p * inv;
    float s, c; __sincosf(ang, &s, &c);
    float x0 = (float)row[j], x1 = (float)row[j + HDIM / 2];
    row[j]            = (_Float16)(x0 * c - x1 * s);
    row[j + HDIM / 2] = (_Float16)(x1 * c + x0 * s);
  }
}

// out = residual + mlp*sigmoid(comb) + att*(1-sigmoid(comb))
__global__ __launch_bounds__(256)
void combine(const float* __restrict__ X, const float* __restrict__ Mlp,
             const float* __restrict__ Att, const float* __restrict__ Comb,
             float* __restrict__ out)
{
  long long i = (long long)blockIdx.x * 256 + threadIdx.x;
  float g = 1.f / (1.f + __expf(-Comb[i]));
  out[i] = X[i] + Mlp[i] * g + Att[i] * (1.f - g);
}

extern "C" void kernel_launch(void* const* d_in, const int* in_sizes, int n_in,
                              void* d_out, int out_size, void* d_ws, size_t ws_size,
                              hipStream_t stream) {
  (void)in_sizes; (void)n_in; (void)out_size; (void)ws_size;
  const float*     X     = (const float*)d_in[0];
  const long long* pos   = (const long long*)d_in[1];
  const float*     wln   = (const float*)d_in[2];
  const float*     Wk1   = (const float*)d_in[3];
  const float*     Wk2   = (const float*)d_in[4];
  const float*     Wv    = (const float*)d_in[5];
  const float*     Wg    = (const float*)d_in[6];
  const float*     Wu    = (const float*)d_in[7];
  const float*     Wd    = (const float*)d_in[8];
  const float*     Wcomb = (const float*)d_in[9];
  float* out = (float*)d_out;

  // Workspace carve-up (all buffers fully rewritten every call -> deterministic).
  char* wsb = (char*)d_ws;
  size_t off = 0;
  auto take = [&](size_t bytes) -> char* {
    char* p = wsb + off; off += (bytes + 255) & ~(size_t)255; return p;
  };
  _Float16* Xn  = (_Float16*)take((size_t)NTOK * HDIM * 2);
  _Float16* Qh  = (_Float16*)take((size_t)NTOK * HDIM * 2);
  _Float16* K1h = (_Float16*)take((size_t)NTOK * HDIM * 2);
  _Float16* K2h = (_Float16*)take((size_t)NTOK * HDIM * 2);
  _Float16* Vt  = (_Float16*)take((size_t)BATCH * HDIM * SEQ * 2);  // V transposed per batch
  float*    Cmb = (float*)take((size_t)NTOK * HDIM * 4);
  float*    Gr  = (float*)take((size_t)NTOK * IDIM * 4);
  _Float16* Hm  = (_Float16*)take((size_t)NTOK * IDIM * 2);
  float*    S1  = (float*)take((size_t)BATCH * SEQ * SEQ * 4);
  _Float16* Ph  = (_Float16*)take((size_t)BATCH * SEQ * SEQ * 2);
  float*    Atf = (float*)take((size_t)NTOK * HDIM * 4);
  float*    Mlf = (float*)take((size_t)NTOK * HDIM * 4);
  // f16 copies of the weights (so every GEMM operand can use async-to-LDS staging)
  _Float16* hWk1 = (_Float16*)take((size_t)HDIM * HDIM * 2);
  _Float16* hWk2 = (_Float16*)take((size_t)HDIM * HDIM * 2);
  _Float16* hWv  = (_Float16*)take((size_t)HDIM * HDIM * 2);
  _Float16* hWcb = (_Float16*)take((size_t)HDIM * HDIM * 2);
  _Float16* hWg  = (_Float16*)take((size_t)IDIM * HDIM * 2);
  _Float16* hWu  = (_Float16*)take((size_t)IDIM * HDIM * 2);
  _Float16* hWd  = (_Float16*)take((size_t)HDIM * IDIM * 2);

  const float inv_sqrt = 1.0f / sqrtf((float)HDIM);
  const dim3 blk(256);
  const dim3 gH(HDIM / BN, NTOK / BM, 1);
  const dim3 gI(IDIM / BN, NTOK / BM, 1);
  const dim3 gS(SEQ / BN, SEQ / BM, BATCH);
  const dim3 gA(HDIM / BN, SEQ / BM, BATCH);
  const long long zSH = (long long)SEQ * HDIM;
  const long long zSS = (long long)SEQ * SEQ;
  const long long zHS = (long long)HDIM * SEQ;

  // 0. Weight precision conversion
  const long long nHH4 = (long long)HDIM * HDIM / 4;
  const long long nIH4 = (long long)IDIM * HDIM / 4;
  cvt_f32_f16<<<(unsigned)((nHH4 + 255) / 256), blk, 0, stream>>>(Wk1, hWk1, nHH4);
  cvt_f32_f16<<<(unsigned)((nHH4 + 255) / 256), blk, 0, stream>>>(Wk2, hWk2, nHH4);
  cvt_f32_f16<<<(unsigned)((nHH4 + 255) / 256), blk, 0, stream>>>(Wv,  hWv,  nHH4);
  cvt_f32_f16<<<(unsigned)((nHH4 + 255) / 256), blk, 0, stream>>>(Wcomb, hWcb, nHH4);
  cvt_f32_f16<<<(unsigned)((nIH4 + 255) / 256), blk, 0, stream>>>(Wg,  hWg,  nIH4);
  cvt_f32_f16<<<(unsigned)((nIH4 + 255) / 256), blk, 0, stream>>>(Wu,  hWu,  nIH4);
  cvt_f32_f16<<<(unsigned)((nIH4 + 255) / 256), blk, 0, stream>>>(Wd,  hWd,  nIH4);

  // 1. RMSNorm + query RoPE
  rmsnorm_rope_q<<<NTOK, blk, 0, stream>>>(X, pos, wln, Xn, Qh);

  // 2. Projections
  gemm_wmma<EP_F16><<<gH, blk, 0, stream>>>(Xn, hWk1, K1h, nullptr,
      NTOK, HDIM, HDIM, inv_sqrt, 0, 0, 0, 0, 0);
  rope_k1<<<NTOK, blk, 0, stream>>>(K1h, pos);
  gemm_wmma<EP_F16><<<gH, blk, 0, stream>>>(Xn, hWk2, K2h, nullptr,
      NTOK, HDIM, HDIM, inv_sqrt, 0, 0, 0, 0, 0);
  gemm_wmma<EP_TRANS><<<gH, blk, 0, stream>>>(Xn, hWv, Vt, nullptr,
      NTOK, HDIM, HDIM, inv_sqrt, 0, 0, 0, 0, 0);
  gemm_wmma<EP_F32><<<gH, blk, 0, stream>>>(Xn, hWcb, Cmb, nullptr,
      NTOK, HDIM, HDIM, 1.0f, 0, 0, 0, 0, 0);

  // 3. MLP: Graw = Xn@Wg^T ; Hmid = silu(Graw) * (Xn@Wu^T) ; Mlp = Hmid@Wd^T
  gemm_wmma<EP_F32><<<gI, blk, 0, stream>>>(Xn, hWg, Gr, nullptr,
      NTOK, IDIM, HDIM, 1.0f, 0, 0, 0, 0, 0);
  gemm_wmma<EP_SILU><<<gI, blk, 0, stream>>>(Xn, hWu, Hm, Gr,
      NTOK, IDIM, HDIM, 1.0f, 0, 0, 0, 0, 0);
  gemm_wmma<EP_F32><<<gH, blk, 0, stream>>>(Hm, hWd, Mlf, nullptr,
      NTOK, HDIM, IDIM, 1.0f, 0, 0, 0, 0, 0);

  // 4. Attention: S1 = Q@K1^T ; P = causal*silu(S1)*(Q@K2^T) ; Att = P@V
  gemm_wmma<EP_F32><<<gS, blk, 0, stream>>>(Qh, K1h, S1, nullptr,
      SEQ, SEQ, HDIM, 1.0f, zSH, zSH, zSS, 0, 1);
  gemm_wmma<EP_CAUSAL><<<gS, blk, 0, stream>>>(Qh, K2h, Ph, S1,
      SEQ, SEQ, HDIM, 1.0f, zSH, zSH, zSS, zSS, 1);
  gemm_wmma<EP_F32><<<gA, blk, 0, stream>>>(Ph, Vt, Atf, nullptr,
      SEQ, HDIM, SEQ, 1.0f, zSS, zHS, zSH, 0, 0);

  // 5. Gated residual combine
  combine<<<(NTOK * HDIM) / 256, blk, 0, stream>>>(X, Mlf, Atf, Cmb, out);
}
